// Proposed_Module_GlobalGNN_4638564679952
// MI455X (gfx1250) — matrix operand order repaired and checked by reading, not compiled, not run on hardware
//
#include <hip/hip_runtime.h>
#include <hip/hip_bf16.h>
#include <cstdint>

// ---------------------------------------------------------------------------
// MI455X / gfx1250 implementation.
// Two batched GEMMs over 512^2 pairwise rows -> v_wmma_f32_16x16x32_f16.
// BatchNorm over the full 512^2 batch forces stats passes (3-pass recompute).
// Round 2: 8-wave workgroups; packed weights staged once per workgroup into
// LDS via the Tensor Data Mover (tensor_load_to_lds + s_wait_tensorcnt), so
// B-fragments are fed from LDS instead of re-streaming L2 per wave.
// ---------------------------------------------------------------------------

typedef __attribute__((ext_vector_type(16))) _Float16 v16h;
typedef __attribute__((ext_vector_type(8)))  float    v8f;
typedef __attribute__((ext_vector_type(4)))  unsigned v4u;
typedef __attribute__((ext_vector_type(8)))  unsigned v8u;

#define NNODE 512
#define C0 96
#define C1 192
#define C2 288
#define EPSBN 1e-5f

#define NW1H 18432   // w1 packed halves: (192/16)*(96/32)*32*16
#define NW2H 55296   // w2 packed halves: (288/16)*(192/32)*32*16
#define NWKH (NW1H + NW2H)   // 73728 halves = 147456 B (contiguous in ws)

// ---------------------------------------------------------------------------
// 0) zero the stat accumulators (workspace is poisoned 0xAA by harness)
// ---------------------------------------------------------------------------
__global__ void k_init(float* st1, float* st2, float* st3) {
  int t = threadIdx.x;
  for (int k = t; k < 2 * C1; k += 256) st1[k] = 0.f;
  for (int k = t; k < 2 * C2; k += 256) st2[k] = 0.f;
  for (int k = t; k < 2 * C0; k += 256) st3[k] = 0.f;
}

// ---------------------------------------------------------------------------
// 1) 4x4x4 window mean: x_concat [1,32,32,32,96] -> x [512,96]
// ---------------------------------------------------------------------------
__global__ void k_window_mean(const float* __restrict__ xc, float* __restrict__ x) {
  int t = blockIdx.x * blockDim.x + threadIdx.x;
  if (t >= NNODE * C0) return;
  int c = t % C0, n = t / C0;
  int gd = n % 8, gw = (n / 8) % 8, gh = n / 64;
  float s = 0.f;
  for (int fh = 0; fh < 4; ++fh)
    for (int fw = 0; fw < 4; ++fw)
      for (int fd = 0; fd < 4; ++fd)
        s += xc[(((gh * 4 + fh) * 32 + (gw * 4 + fw)) * 32 + (gd * 4 + fd)) * C0 + c];
  x[t] = s * (1.f / 64.f);
}

// ---------------------------------------------------------------------------
// 2) pack w1 (192x96) and w2 (288x192) into WMMA B-fragment-linear f16 order.
//    B 32x16 f16 layout (ISA 7.12.2): lane l holds column N=l%16;
//    half v holds K = (v<8 ? v : v+8) + (l>=16 ? 8 : 0).
//    Packed offset = ((nt*KT + kt)*32 + lane)*16 + v;  w2 block follows w1.
// ---------------------------------------------------------------------------
__global__ void k_pack(const float* __restrict__ w1, const float* __restrict__ w2,
                       _Float16* __restrict__ wpk) {
  int t = blockIdx.x * blockDim.x + threadIdx.x;
  if (t < NW1H) {
    int v = t & 15, l = (t >> 4) & 31, q = t >> 9;
    int kt = q % 3, nt = q / 3;
    int o = nt * 16 + (l & 15);
    int c = kt * 32 + (v < 8 ? v : v + 8) + ((l & 16) ? 8 : 0);
    wpk[t] = (_Float16)w1[o * C0 + c];
  } else if (t < NWKH) {
    int u = t - NW1H;
    int v = u & 15, l = (u >> 4) & 31, q = u >> 9;
    int kt = q % 6, nt = q / 6;
    int o = nt * 16 + (l & 15);
    int c = kt * 32 + (v < 8 ? v : v + 8) + ((l & 16) ? 8 : 0);
    wpk[t] = (_Float16)w2[o * C1 + c];
  }
}

// ---------------------------------------------------------------------------
// 3) fused pairwise-GEMM kernel. 256 threads = 8 waves per workgroup.
//    Workgroup: fixed i, 128-wide j block; wave wv owns 16 rows
//    d[i, j0+16wv .. +15, :] = |x_i - x_j|.
//    Packed weights (147456 B) are TDM-staged into LDS once per workgroup.
//    PASS 1: GEMM1 raw -> per-channel sum/sumsq (BN1 stats)
//    PASS 2: GEMM1+BN1+lrelu -> GEMM2 raw -> BN2 stats
//    PASS 3: full chain -> dot(w3) -> logits a[i,j] (diag masked)
//    st layout: [sum | sumsq | sc | sh]; sc = gamma*rsqrt(var+eps),
//               sh = beta - mean*sc.
// ---------------------------------------------------------------------------
template <int PASS>
__global__ __launch_bounds__(256) void k_gnn(
    const float* __restrict__ x, const _Float16* __restrict__ wpk,
    const float* __restrict__ b1, const float* __restrict__ b2,
    const float* __restrict__ w3, const float* __restrict__ b3,
    float* __restrict__ st1, float* __restrict__ st2,
    float* __restrict__ aout, int use_tdm) {
  __shared__ __attribute__((aligned(32))) _Float16 bsm[NWKH];   // 147456 B
  __shared__ float    xi[C0];                                    //    384 B
  __shared__ float    xj[128 * C0];                              //  49152 B
  __shared__ _Float16 h1t[8][16 * C1];                           //  49152 B
  __shared__ float    redA[C2];                                  //   1152 B
  __shared__ float    redB[C2];                                  //   1152 B
  __shared__ float    arow[128];                                 //    512 B

  const int tid  = threadIdx.x;
  const int lane = tid & 31;
  const int wv   = tid >> 5;
  const int i    = blockIdx.x >> 2;          // 0..511
  const int j0   = (blockIdx.x & 3) << 7;    // 128-wide j block
  const int jb   = j0 + wv * 16;             // this wave's 16 rows
  const int m16  = lane & 15;
  const int hi   = (lane & 16) ? 8 : 0;      // A-frag / D-frag half select

  // cooperative stage of x rows (global -> LDS)
  for (int t = tid; t < C0; t += 256) xi[t] = x[i * C0 + t];
  for (int t = tid; t < 128 * C0; t += 256) xj[t] = x[(j0 + t / C0) * C0 + (t % C0)];
  if (PASS == 1 || PASS == 2)
    for (int t = tid; t < C2; t += 256) { redA[t] = 0.f; redB[t] = 0.f; }

  // --- Tensor Data Mover: packed weights (global, L2-resident) -> LDS ---
  if (use_tdm) {
    if (wv == 0) {
      unsigned lds_b = (unsigned)(uintptr_t)&bsm[0];          // flat low 32b = LDS byte addr
      unsigned long long ga = (unsigned long long)(uintptr_t)wpk;
      const unsigned NE = NWKH * 2u / 8u;                     // 18432 x 8-byte elements
      v4u g0; v8u g1;
      g0[0] = 1u;                                             // count=1, user, no gather
      g0[1] = lds_b;                                          // D#.lds_addr
      g0[2] = (unsigned)ga;                                   // global_addr[31:0]
      g0[3] = (unsigned)((ga >> 32) & 0x01FFFFFFu) | (2u << 30); // addr[56:32] | type=2
      g1[0] = 3u << 16;                                       // data_size = 8B, mask=0
      g1[1] = (NE & 0xFFFFu) << 16;                           // tensor_dim0[15:0]
      g1[2] = (NE >> 16) | (1u << 16);                        // tensor_dim0[31:16] | dim1=1
      g1[3] = (NE & 0xFFFFu) << 16;                           // tile_dim0
      g1[4] = 1u;                                             // tile_dim1=1, tile_dim2=0
      g1[5] = NE;                                             // tensor_dim0_stride lo
      g1[6] = 0u;
      g1[7] = 0u;
      asm volatile("tensor_load_to_lds %0, %1" :: "s"(g0), "s"(g1) : "memory");
      __builtin_amdgcn_s_wait_tensorcnt(0);
    }
  } else {
    for (int t = tid; t < NWKH; t += 256) bsm[t] = wpk[t];    // fallback path
  }
  __syncthreads();

  // --- A1 fragments: 16x32 f16, A[m][k] = |x_i[k] - x_{jb+m}[k]| ---
  // A layout (ISA 7.12.2): lane holds row M = lane%16;
  // half v -> K = (v<8 ? v : v+8) + (lane>=16 ? 8 : 0)  (+ 32*kt)
  v16h a1[3];
#pragma unroll
  for (int kt = 0; kt < 3; ++kt)
#pragma unroll
    for (int v = 0; v < 16; ++v) {
      int kk = kt * 32 + (v < 8 ? v : v + 8) + hi;
      a1[kt][v] = (_Float16)__builtin_fabsf(xi[kk] - xj[(wv * 16 + m16) * C0 + kk]);
    }

  // ---------------- GEMM1: [16x96] x [96x192], B from LDS ----------------
  for (int nt = 0; nt < C1 / 16; ++nt) {
    v8f c = {};
#pragma unroll
    for (int kt = 0; kt < 3; ++kt) {
      v16h b = *(const v16h*)(&bsm[((nt * 3 + kt) * 32 + lane) * 16]);
      c = __builtin_amdgcn_wmma_f32_16x16x32_f16(false, a1[kt], false, b,
                                                 (short)0, c, false, false);
    }
    // D layout: VGPR r -> row M = r + hi; channel N = lane%16 (fixed per lane)
    int ch = nt * 16 + m16;
    float bias = b1[ch];
    if (PASS == 1) {
      float s = 0.f, s2 = 0.f;
#pragma unroll
      for (int r = 0; r < 8; ++r) { float hv = c[r] + bias; s += hv; s2 += hv * hv; }
      atomicAdd(&redA[ch], s);
      atomicAdd(&redB[ch], s2);
    } else {
      float sc = st1[2 * C1 + ch], sh = st1[3 * C1 + ch];
#pragma unroll
      for (int r = 0; r < 8; ++r) {
        float hv = (c[r] + bias) * sc + sh;
        hv = hv > 0.f ? hv : 0.2f * hv;                 // leaky_relu(0.2)
        h1t[wv][(r + hi) * C1 + ch] = (_Float16)hv;     // wave-private f16 tile
      }
    }
  }

  if (PASS == 1) {
    __syncthreads();
    for (int t = tid; t < C1; t += 256) {
      atomicAdd(&st1[t], redA[t]);
      atomicAdd(&st1[C1 + t], redB[t]);
    }
    return;
  }
  __syncthreads();

  // --- D->A relayout through LDS: A2 fragments for K=192 ---
  v16h a2[6];
#pragma unroll
  for (int kt = 0; kt < 6; ++kt)
#pragma unroll
    for (int v = 0; v < 16; ++v) {
      int kk = kt * 32 + (v < 8 ? v : v + 8) + hi;
      a2[kt][v] = h1t[wv][m16 * C1 + kk];
    }

  float arow_acc[8];
#pragma unroll
  for (int r = 0; r < 8; ++r) arow_acc[r] = 0.f;

  // ---------------- GEMM2: [16x192] x [192x288], B from LDS ----------------
  for (int nt = 0; nt < C2 / 16; ++nt) {
    v8f c = {};
#pragma unroll
    for (int kt = 0; kt < 6; ++kt) {
      v16h b = *(const v16h*)(&bsm[NW1H + ((nt * 6 + kt) * 32 + lane) * 16]);
      c = __builtin_amdgcn_wmma_f32_16x16x32_f16(false, a2[kt], false, b,
                                                 (short)0, c, false, false);
    }
    int ch = nt * 16 + m16;
    float bias = b2[ch];
    if (PASS == 2) {
      float s = 0.f, s2 = 0.f;
#pragma unroll
      for (int r = 0; r < 8; ++r) { float hv = c[r] + bias; s += hv; s2 += hv * hv; }
      atomicAdd(&redA[ch], s);
      atomicAdd(&redB[ch], s2);
    } else {
      float sc = st2[2 * C2 + ch], sh = st2[3 * C2 + ch], wvw = w3[ch];
#pragma unroll
      for (int r = 0; r < 8; ++r) {
        float hv = (c[r] + bias) * sc + sh;
        hv = hv > 0.f ? hv : 0.2f * hv;
        arow_acc[r] += hv * wvw;                        // partial dot with w3
      }
    }
  }

  if (PASS == 2) {
    __syncthreads();
    for (int t = tid; t < C2; t += 256) {
      atomicAdd(&st2[t], redA[t]);
      atomicAdd(&st2[C2 + t], redB[t]);
    }
    return;
  }

  // PASS 3: reduce per-row logits across the 16 lanes sharing each row
  if (tid < 128) arow[tid] = 0.f;
  __syncthreads();
#pragma unroll
  for (int r = 0; r < 8; ++r) atomicAdd(&arow[wv * 16 + r + hi], arow_acc[r]);
  __syncthreads();
  if (lane < 16) {
    int j = jb + lane;
    float av = arow[wv * 16 + lane] + b3[0];
    if (j == i) av -= 1e8f;                             // mask diagonal
    aout[(size_t)i * NNODE + j] = av;
  }
}

// ---------------------------------------------------------------------------
// 4) finalize BN stats: sum/sumsq -> (scale, shift)
// ---------------------------------------------------------------------------
__global__ void k_finalize(float* st, const float* __restrict__ gamma,
                           const float* __restrict__ beta, int nch, float invcnt) {
  int c = blockIdx.x * blockDim.x + threadIdx.x;
  if (c >= nch) return;
  float mean = st[c] * invcnt;
  float var  = st[nch + c] * invcnt - mean * mean;
  float inv  = rsqrtf(var + EPSBN);
  float sc   = gamma[c] * inv;
  st[2 * nch + c] = sc;
  st[3 * nch + c] = beta[c] - mean * sc;
}

// ---------------------------------------------------------------------------
// 5) row softmax of a[512,512] + aggregation aggS[i,:] = sum_j S[i,j] x[j,:]
// ---------------------------------------------------------------------------
__global__ __launch_bounds__(256) void k_softmax_agg(const float* __restrict__ a,
                                                     const float* __restrict__ x,
                                                     float* __restrict__ aggS) {
  __shared__ float red[256];
  __shared__ float p[NNODE];
  int i = blockIdx.x, t = threadIdx.x;
  const float* ar = a + (size_t)i * NNODE;

  float m = -3.4e38f;
  for (int j = t; j < NNODE; j += 256) m = fmaxf(m, ar[j]);
  red[t] = m; __syncthreads();
  for (int s = 128; s > 0; s >>= 1) {
    if (t < s) red[t] = fmaxf(red[t], red[t + s]);
    __syncthreads();
  }
  m = red[0]; __syncthreads();

  float l = 0.f;
  for (int j = t; j < NNODE; j += 256) {
    float e = __expf(ar[j] - m);
    p[j] = e; l += e;
  }
  red[t] = l; __syncthreads();
  for (int s = 128; s > 0; s >>= 1) {
    if (t < s) red[t] += red[t + s];
    __syncthreads();
  }
  float inv = 1.f / red[0]; __syncthreads();
  for (int j = t; j < NNODE; j += 256) p[j] *= inv;
  __syncthreads();

  for (int c = t; c < C0; c += 256) {
    float s = 0.f;
    for (int j = 0; j < NNODE; ++j) s += p[j] * x[j * C0 + c];
    aggS[i * C0 + c] = s;
  }
}

// ---------------------------------------------------------------------------
// 6) GCN: y_raw = leaky_relu([x | aggS] @ gcn_w), accumulate BN3 stats
// ---------------------------------------------------------------------------
__global__ void k_gcn(const float* __restrict__ x, const float* __restrict__ aggS,
                      const float* __restrict__ gcn_w, float* st3,
                      float* __restrict__ y_raw) {
  int t = blockIdx.x * blockDim.x + threadIdx.x;
  if (t >= NNODE * C0) return;
  int n = t / C0, c = t % C0;
  float s = 0.f;
  for (int k = 0; k < C0; ++k) s += x[n * C0 + k] * gcn_w[k * C0 + c];
  for (int k = 0; k < C0; ++k) s += aggS[n * C0 + k] * gcn_w[(C0 + k) * C0 + c];
  s = s > 0.f ? s : 0.2f * s;
  y_raw[t] = s;
  atomicAdd(&st3[c], s);
  atomicAdd(&st3[C0 + c], s * s);
}

// ---------------------------------------------------------------------------
// 7) unfold-upsample + BN3 affine + residual:
//    out[h,w,d,c] = x_concat + xnew_padded[h/4+h%4-1, w/4+w%4-1, d/4+d%4-1, c]
// ---------------------------------------------------------------------------
__global__ void k_output(const float* __restrict__ xc, const float* __restrict__ y_raw,
                         const float* __restrict__ st3, float* __restrict__ out) {
  int t = blockIdx.x * blockDim.x + threadIdx.x;
  if (t >= 32 * 32 * 32 * C0) return;
  int c = t % C0, rest = t / C0;
  int d = rest % 32; rest /= 32;
  int w = rest % 32; int h = rest / 32;
  int gh = (h >> 2) + (h & 3) - 1;
  int gw = (w >> 2) + (w & 3) - 1;
  int gd = (d >> 2) + (d & 3) - 1;
  float add = 0.f;
  if ((unsigned)gh < 8u && (unsigned)gw < 8u && (unsigned)gd < 8u) {
    int n = (gh * 8 + gw) * 8 + gd;
    add = y_raw[n * C0 + c] * st3[2 * C0 + c] + st3[3 * C0 + c];
  }
  out[t] = xc[t] + add;
}

// ---------------------------------------------------------------------------
extern "C" void kernel_launch(void* const* d_in, const int* in_sizes, int n_in,
                              void* d_out, int out_size, void* d_ws, size_t ws_size,
                              hipStream_t stream) {
  const float* x_concat = (const float*)d_in[0];
  const float* w1   = (const float*)d_in[1];
  const float* b1   = (const float*)d_in[2];
  const float* g1   = (const float*)d_in[3];
  const float* be1  = (const float*)d_in[4];
  const float* w2   = (const float*)d_in[5];
  const float* b2   = (const float*)d_in[6];
  const float* g2   = (const float*)d_in[7];
  const float* be2  = (const float*)d_in[8];
  const float* w3   = (const float*)d_in[9];
  const float* b3   = (const float*)d_in[10];
  const float* gcnw = (const float*)d_in[11];
  const float* g3   = (const float*)d_in[12];
  const float* be3  = (const float*)d_in[13];
  float* out = (float*)d_out;

  // ---- workspace layout (all 32B-aligned; total ~1.72 MB) ----
  uintptr_t base = (uintptr_t)d_ws;
  float*    x    = (float*)(base + 0);              // 512*96 f32   = 196608 B
  _Float16* wpk  = (_Float16*)(base + 196608);      // 73728 f16    = 147456 B (w1p|w2p)
  float*    st1  = (float*)(base + 344064);         // 4*192 f32    =   3072 B
  float*    st2  = (float*)(base + 347136);         // 4*288 f32    =   4608 B
  float*    st3  = (float*)(base + 351744);         // 4*96  f32    =   1536 B
  float*    amat = (float*)(base + 353280);         // 512*512 f32  = 1048576 B
  float*    aggS = (float*)(base + 1401856);        // 512*96 f32   = 196608 B
  float*    yraw = (float*)(base + 1598464);        // 512*96 f32   = 196608 B

  const float invN2 = 1.f / (float)(NNODE * (size_t)NNODE);  // BN over 512^2
  const float invN  = 1.f / (float)NNODE;                    // BN over 512 nodes
  const int use_tdm = 1;

  k_init<<<1, 256, 0, stream>>>(st1, st2, st3);
  k_window_mean<<<(NNODE * C0 + 255) / 256, 256, 0, stream>>>(x_concat, x);
  k_pack<<<(NWKH + 255) / 256, 256, 0, stream>>>(w1, w2, wpk);

  dim3 gnn_grid(NNODE * 4);  // 512 i * 4 j-blocks of 128
  k_gnn<1><<<gnn_grid, 256, 0, stream>>>(x, wpk, b1, b2, w3, b3, st1, st2, amat, use_tdm);
  k_finalize<<<1, 256, 0, stream>>>(st1, g1, be1, C1, invN2);
  k_gnn<2><<<gnn_grid, 256, 0, stream>>>(x, wpk, b1, b2, w3, b3, st1, st2, amat, use_tdm);
  k_finalize<<<2, 256, 0, stream>>>(st2, g2, be2, C2, invN2);
  k_gnn<3><<<gnn_grid, 256, 0, stream>>>(x, wpk, b1, b2, w3, b3, st1, st2, amat, use_tdm);

  k_softmax_agg<<<NNODE, 256, 0, stream>>>(amat, x, aggS);
  k_gcn<<<(NNODE * C0 + 255) / 256, 256, 0, stream>>>(x, aggS, gcnw, st3, yraw);
  k_finalize<<<1, 256, 0, stream>>>(st3, g3, be3, C0, invN);
  k_output<<<(32 * 32 * 32 * C0 + 255) / 256, 256, 0, stream>>>(x_concat, yraw, st3, out);
}
